// LSTMTimeSeriesPredictor_13426067767400
// MI455X (gfx1250) — compile-verified, matrix operand order checked by reading, not run on hardware
//
#include <hip/hip_runtime.h>

typedef __attribute__((ext_vector_type(16))) _Float16 v16h;
typedef __attribute__((ext_vector_type(8)))  _Float16 v8h;
typedef __attribute__((ext_vector_type(8)))  float    v8f;

#define NB   64    // batch
#define NT   512   // encoder time steps
#define NI   64    // input dim
#define NH   512   // hidden dim
#define NO   64    // output dim
#define NTGT 96    // decoder steps
#define NG   2048  // 4*NH gates

#define K0CAT 576   // NI + NH  (layer0 concat K)
#define K1CAT 1024  // NH + NH  (layer1 concat K)
#define KC0   18    // 576/32
#define KC1   32    // 1024/32
#define KCFC  16    // 512/32

#define NWG  16
#define TPB  256    // 8 waves; 16*8 = 128 waves = 4 mtiles x 32 htiles

// ---------------------------------------------------------------------------
// Device-wide software barrier (16 resident WGs). Monotonic counter, reset by
// init kernel every launch -> graph-replay safe.
// ---------------------------------------------------------------------------
__device__ __forceinline__ void grid_barrier(unsigned* bar, unsigned* gen) {
  __threadfence();
  __syncthreads();
  if (threadIdx.x == 0) {
    unsigned target = *gen + gridDim.x;
    __atomic_fetch_add(bar, 1u, __ATOMIC_ACQ_REL);
    while (__atomic_load_n(bar, __ATOMIC_ACQUIRE) < target) {
      __builtin_amdgcn_s_sleep(1);
    }
  }
  *gen += gridDim.x;
  __syncthreads();
  __threadfence();
}

// ---------------------------------------------------------------------------
// WMMA fragment loaders (ISA 7.12.2 16-bit A layout):
// lane L: row m = L%16, hh = L/16; elems 0..7 -> K = kbase+8*hh+e,
// elems 8..15 -> K = kbase+16+8*hh+(e-8)  => two contiguous 16B loads.
// ---------------------------------------------------------------------------
__device__ __forceinline__ v16h frag_A_lds(const _Float16* sA, int kbase, int krow) {
  int lane = threadIdx.x & 31;
  int hh = lane >> 4;
  const _Float16* p = sA + (lane & 15) * krow + kbase + 8 * hh;
  v8h lo = *(const v8h*)p;
  v8h hi = *(const v8h*)(p + 16);
  v16h r;
#pragma unroll
  for (int e = 0; e < 8; ++e) { r[e] = lo[e]; r[e + 8] = hi[e]; }
  return r;
}

__device__ __forceinline__ v16h frag_A_glb(const _Float16* __restrict__ A,
                                           int mbase, int kbase, int stride) {
  int lane = threadIdx.x & 31;
  int hh = lane >> 4;
  const _Float16* p = A + (size_t)(mbase + (lane & 15)) * stride + kbase + 8 * hh;
  v8h lo = *(const v8h*)p;
  v8h hi = *(const v8h*)(p + 16);
  v16h r;
#pragma unroll
  for (int e = 0; e < 8; ++e) { r[e] = lo[e]; r[e + 8] = hi[e]; }
  return r;
}

// B pre-packed in exact lane order: 16 contiguous halves per lane per chunk.
__device__ __forceinline__ v16h frag_B(const _Float16* __restrict__ Wp,
                                       int ntile, int kc, int kchunks) {
  int lane = threadIdx.x & 31;
  const _Float16* p = Wp + (((size_t)ntile * kchunks + kc) * 32 + lane) * 16;
  v8h lo = *(const v8h*)p;
  v8h hi = *(const v8h*)(p + 8);
  v16h r;
#pragma unroll
  for (int e = 0; e < 8; ++e) { r[e] = lo[e]; r[e + 8] = hi[e]; }
  return r;
}

// ---------------------------------------------------------------------------
// Cooperative stage of one 16-row A panel into LDS (all 8 waves share mtile).
// ---------------------------------------------------------------------------
template <int KROW>
__device__ __forceinline__ void stage_panel(const _Float16* Asrc,
                                            int mbase, _Float16* sA) {
  constexpr int chunks = (16 * KROW) / 8;  // v8h chunks
  for (int i = threadIdx.x; i < chunks; i += TPB) {
    int h0 = i * 8;
    int row = h0 / KROW;
    int col = h0 - row * KROW;
    *(v8h*)(sA + h0) = *(const v8h*)(Asrc + (size_t)(mbase + row) * KROW + col);
  }
  __syncthreads();
}

// ---------------------------------------------------------------------------
// 4-gate GEMM: acc[g] += A(16xK) * Wp[gate g tile] ; double-buffered K loop.
// Wp carries an opaque (loop-variant) offset so the compiler cannot hoist the
// weight fragments across time steps (which spilled >1024 VGPRs to scratch);
// weights intentionally stream from L2 every step as global_load_b128.
// ---------------------------------------------------------------------------
template <int KC, int KROW>
__device__ __forceinline__ void gemm_gates4(const _Float16* sA,
                                            const _Float16* __restrict__ Wp,
                                            int htile, v8f acc[4]) {
  v16h a_cur = frag_A_lds(sA, 0, KROW);
  v16h b_cur[4];
#pragma unroll
  for (int g = 0; g < 4; ++g) b_cur[g] = frag_B(Wp, g * 32 + htile, 0, KC);
#pragma unroll
  for (int kc = 0; kc < KC; ++kc) {
    const int kn = (kc + 1 < KC) ? kc + 1 : kc;
    v16h a_nxt = frag_A_lds(sA, kn * 32, KROW);
    v16h b_nxt[4];
#pragma unroll
    for (int g = 0; g < 4; ++g) b_nxt[g] = frag_B(Wp, g * 32 + htile, kn, KC);
#pragma unroll
    for (int g = 0; g < 4; ++g)
      acc[g] = __builtin_amdgcn_wmma_f32_16x16x32_f16(
          false, a_cur, false, b_cur[g], (short)0, acc[g], false, false);
    a_cur = a_nxt;
#pragma unroll
    for (int g = 0; g < 4; ++g) b_cur[g] = b_nxt[g];
  }
}

__device__ __forceinline__ float sigm(float x) { return 1.f / (1.f + __expf(-x)); }

// In-register LSTM cell update; cell state cs[8] stays in VGPRs across steps.
__device__ __forceinline__ void act_store(const v8f acc[4], const float bias[4],
                                          float cs[8], int mtile, int htile,
                                          _Float16* hDst0, int stride0, int off0,
                                          _Float16* hDst1, int stride1, int off1) {
  int lane = threadIdx.x & 31;
  int h = htile * 16 + (lane & 15);
  int mb = mtile * 16 + 8 * (lane >> 4);
#pragma unroll
  for (int r = 0; r < 8; ++r) {
    float iv = acc[0][r] + bias[0];
    float fv = acc[1][r] + bias[1];
    float gv = acc[2][r] + bias[2];
    float ov = acc[3][r] + bias[3];
    float cn = sigm(fv) * cs[r] + sigm(iv) * tanhf(gv);
    float hn = sigm(ov) * tanhf(cn);
    cs[r] = cn;
    _Float16 h16 = (_Float16)hn;
    hDst0[(size_t)(mb + r) * stride0 + off0 + h] = h16;
    if (hDst1) hDst1[(size_t)(mb + r) * stride1 + off1 + h] = h16;
  }
}

// ---------------------------------------------------------------------------
// Persistent fused kernel: 512 encoder + 96 decoder steps.
// 128 waves; each wave owns one (mtile, htile) 16x16 range incl. all 4 gates,
// its cell state (c0,c1) in registers. 2 grid barriers/step (3 in decoder).
// Ping-pong A0/A1 buffers remove intra-phase read/write races.
// ---------------------------------------------------------------------------
__global__ void __launch_bounds__(TPB)
lstm_main(const _Float16* __restrict__ wp0_in, const _Float16* __restrict__ wp1_in,
          const _Float16* __restrict__ wpfc_in, const _Float16* __restrict__ x16,
          _Float16* A0, _Float16* A1,
          const float* __restrict__ bih0, const float* __restrict__ bhh0,
          const float* __restrict__ bih1, const float* __restrict__ bhh1,
          const float* __restrict__ bfc, float* __restrict__ out,
          unsigned* bar) {
  __shared__ __align__(16) _Float16 sA[16 * K1CAT];  // 32 KB panel

  const int wave = threadIdx.x >> 5;
  const int gwave = blockIdx.x * (TPB / 32) + wave;  // 0..127
  const int mtile = gwave >> 5;                      // 0..3 (uniform per block)
  const int htile = gwave & 31;                      // 0..31
  const int lane = threadIdx.x & 31;
  const int h = htile * 16 + (lane & 15);

  // biases folded into per-lane registers once
  float bias0[4], bias1[4];
#pragma unroll
  for (int g = 0; g < 4; ++g) {
    bias0[g] = bih0[g * NH + h] + bhh0[g * NH + h];
    bias1[g] = bih1[g * NH + h] + bhh1[g * NH + h];
  }
  // cell state lives in VGPRs for the entire kernel
  float c0s[8] = {0, 0, 0, 0, 0, 0, 0, 0};
  float c1s[8] = {0, 0, 0, 0, 0, 0, 0, 0};
  const v8f vzero = {};
  unsigned gen = 0;

  for (int step = 0; step < NT + NTGT; ++step) {
    // Opaque zero offset, redefined every iteration: keeps pointer provenance
    // (-> global_load, LOADcnt only) while making the weight addresses
    // loop-variant so fragments cannot be hoisted across steps (no spills).
    size_t wofs = 0;
    asm volatile("" : "+s"(wofs));
    const _Float16* __restrict__ wp0 = wp0_in + wofs;
    const _Float16* __restrict__ wp1 = wp1_in + wofs;
    const _Float16* __restrict__ wpfc = wpfc_in + wofs;

    const int p = step & 1;
    const _Float16* A0r = A0 + (size_t)p * NB * K0CAT;        // read:  x_t | h0_{t-1}
    _Float16* A0w = A0 + (size_t)(1 - p) * NB * K0CAT;        // write: x_{t+1} | h0_t
    _Float16* A1r = A1 + (size_t)p * NB * K1CAT;              // read:  h0_t | h1_{t-1}
    _Float16* A1w = A1 + (size_t)(1 - p) * NB * K1CAT;        // write: h1_t

    // ---- phase 0: layer0 gates + cell update ----
    stage_panel<K0CAT>(A0r, mtile * 16, sA);
    {
      v8f acc[4] = {vzero, vzero, vzero, vzero};
      gemm_gates4<KC0, K0CAT>(sA, wp0, htile, acc);
      act_store(acc, bias0, c0s, mtile, htile,
                A1r, K1CAT, 0,    // h0_t -> A1[p][:, 0:512]   (layer1 input, this step)
                A0w, K0CAT, NI);  // h0_t -> A0[1-p][:, 64:576] (layer0 input, next step)
    }
    grid_barrier(bar, &gen);

    // ---- phase 1: layer1 gates + cell update (+ encoder x staging) ----
    stage_panel<K1CAT>(A1r, mtile * 16, sA);
    {
      v8f acc[4] = {vzero, vzero, vzero, vzero};
      gemm_gates4<KC1, K1CAT>(sA, wp1, htile, acc);
      act_store(acc, bias1, c1s, mtile, htile,
                A1w, K1CAT, NH, nullptr, 0, 0);  // h1_t -> A1[1-p][:, 512:1024]
    }
    if (step < NT) {  // stage x_{t+1} (last step keeps x[:,T-1,:] for decoder)
      int g = blockIdx.x * TPB + threadIdx.x;    // exactly NB*NI = 4096 threads
      int tn = (step + 1 < NT) ? (step + 1) : (NT - 1);
      A0w[(size_t)(g >> 6) * K0CAT + (g & 63)] = x16[(size_t)tn * NB * NI + g];
    }
    grid_barrier(bar, &gen);

    // ---- phase 2 (decoder only): y = h1 @ Wfc^T + bfc, feed back as x ----
    if (step >= NT) {
      const int s = step - NT;
      if (gwave < 16) {  // 4x4 tiles, K=512
        const int fmt = gwave >> 2, fnt = gwave & 3;
        v8f acc = vzero;
        for (int kc = 0; kc < KCFC; ++kc) {
          v16h a = frag_A_glb(A1w, fmt * 16, NH + kc * 32, K1CAT);  // h1_t
          v16h b = frag_B(wpfc, fnt, kc, KCFC);
          acc = __builtin_amdgcn_wmma_f32_16x16x32_f16(
              false, a, false, b, (short)0, acc, false, false);
        }
        const int n = fnt * 16 + (lane & 15);
        const int mb = fmt * 16 + 8 * (lane >> 4);
        const float bv = bfc[n];
#pragma unroll
        for (int r = 0; r < 8; ++r) {
          float y = acc[r] + bv;
          int m = mb + r;
          out[(size_t)m * NTGT * NO + (size_t)s * NO + n] = y;  // (B, TGT, O)
          A0w[(size_t)m * K0CAT + n] = (_Float16)y;             // next step's x
        }
      }
      grid_barrier(bar, &gen);
    }
  }
}

// ---------------------------------------------------------------------------
// Init kernels (run each launch; deterministic; graph-capture safe)
// ---------------------------------------------------------------------------
// Pack concat weight [Wih | Whh] (row-major (4H, K0/K1)) into exact per-lane
// WMMA B order: out[((ntile*kchunks + kc)*32 + lane)*16 + e], f16.
__global__ void pack_weights(const float* __restrict__ Wih,
                             const float* __restrict__ Whh,
                             int K0, int K1, int ntiles, int kchunks,
                             _Float16* __restrict__ outp) {
  size_t total = (size_t)ntiles * kchunks * 32 * 16;
  size_t idx = (size_t)blockIdx.x * blockDim.x + threadIdx.x;
  if (idx >= total) return;
  int e = (int)(idx & 15);
  int lane = (int)((idx >> 4) & 31);
  size_t rest = idx >> 9;
  int kc = (int)(rest % kchunks);
  int ntile = (int)(rest / kchunks);
  int n = ntile * 16 + (lane & 15);
  int hh = lane >> 4;
  int kl = (e < 8) ? (e + 8 * hh) : (e + 8 + 8 * hh);
  int k = kc * 32 + kl;
  float v = 0.f;
  if (k < K0)           v = Wih[(size_t)n * K0 + k];
  else if (k - K0 < K1) v = Whh[(size_t)n * K1 + (k - K0)];
  outp[idx] = (_Float16)v;
}

// x (B,T,I) f32 -> x16 [t][b][i] f16 for coalesced per-step staging.
__global__ void pack_x(const float* __restrict__ x, _Float16* __restrict__ x16) {
  size_t idx = (size_t)blockIdx.x * blockDim.x + threadIdx.x;
  if (idx >= (size_t)NB * NT * NI) return;
  int i = (int)(idx % NI);
  size_t r = idx / NI;
  int t = (int)(r % NT);
  int b = (int)(r / NT);
  x16[(size_t)t * NB * NI + (size_t)b * NI + i] = (_Float16)x[idx];
}

__global__ void init_state(_Float16* A0, _Float16* A1,
                           const _Float16* __restrict__ x16, unsigned* bar) {
  int g = blockIdx.x * blockDim.x + threadIdx.x;
  int stride = gridDim.x * blockDim.x;
  // A0[0]: x part <- x[:,0,:], h part <- 0 ; A0[1]: all zero
  for (int i = g; i < 2 * NB * K0CAT; i += stride) {
    int buf = i / (NB * K0CAT);
    int rem = i - buf * (NB * K0CAT);
    int b = rem / K0CAT, cpos = rem - b * K0CAT;
    A0[i] = (buf == 0 && cpos < NI) ? x16[(size_t)b * NI + cpos] : (_Float16)0.f;
  }
  for (int i = g; i < 2 * NB * K1CAT; i += stride) A1[i] = (_Float16)0.f;
  if (g == 0) *bar = 0u;
}

// ---------------------------------------------------------------------------
// Host launcher
// ---------------------------------------------------------------------------
extern "C" void kernel_launch(void* const* d_in, const int* in_sizes, int n_in,
                              void* d_out, int out_size, void* d_ws, size_t ws_size,
                              hipStream_t stream) {
  const float* x     = (const float*)d_in[0];
  const float* W_ih0 = (const float*)d_in[1];
  const float* W_hh0 = (const float*)d_in[2];
  const float* b_ih0 = (const float*)d_in[3];
  const float* b_hh0 = (const float*)d_in[4];
  const float* W_ih1 = (const float*)d_in[5];
  const float* W_hh1 = (const float*)d_in[6];
  const float* b_ih1 = (const float*)d_in[7];
  const float* b_hh1 = (const float*)d_in[8];
  const float* W_fc  = (const float*)d_in[9];
  const float* b_fc  = (const float*)d_in[10];
  float* out = (float*)d_out;

  char* base = (char*)d_ws;
  size_t off = 0;
  auto take = [&](size_t bytes) {
    char* p = base + off;
    off = (off + bytes + 255) & ~(size_t)255;
    return p;
  };
  _Float16* wp0  = (_Float16*)take((size_t)NG * K0CAT * 2);       // 2.25 MB
  _Float16* wp1  = (_Float16*)take((size_t)NG * K1CAT * 2);       // 4 MB
  _Float16* wpfc = (_Float16*)take((size_t)NO * NH * 2);          // 64 KB
  _Float16* x16  = (_Float16*)take((size_t)NB * NT * NI * 2);     // 4 MB
  _Float16* A0   = (_Float16*)take((size_t)2 * NB * K0CAT * 2);   // ping-pong
  _Float16* A1   = (_Float16*)take((size_t)2 * NB * K1CAT * 2);   // ping-pong
  unsigned* bar  = (unsigned*)take(256);
  (void)ws_size; (void)n_in; (void)in_sizes; (void)out_size;

  // phase 1: pack weights + x to f16 WMMA layouts
  {
    size_t tot0 = (size_t)128 * KC0 * 32 * 16;
    pack_weights<<<(unsigned)((tot0 + 255) / 256), 256, 0, stream>>>(
        W_ih0, W_hh0, NI, NH, 128, KC0, wp0);
    size_t tot1 = (size_t)128 * KC1 * 32 * 16;
    pack_weights<<<(unsigned)((tot1 + 255) / 256), 256, 0, stream>>>(
        W_ih1, W_hh1, NH, NH, 128, KC1, wp1);
    size_t totf = (size_t)4 * KCFC * 32 * 16;
    pack_weights<<<(unsigned)((totf + 255) / 256), 256, 0, stream>>>(
        W_fc, W_fc, NH, 0, 4, KCFC, wpfc);
    size_t totx = (size_t)NB * NT * NI;
    pack_x<<<(unsigned)((totx + 255) / 256), 256, 0, stream>>>(x, x16);
  }

  // phase 2: zero/seed activation buffers, reset barrier counter
  init_state<<<128, 256, 0, stream>>>(A0, A1, x16, bar);

  // phase 3: one persistent kernel runs the entire 512+96 step recurrence
  lstm_main<<<NWG, TPB, 0, stream>>>(wp0, wp1, wpfc, x16, A0, A1, b_ih0, b_hh0,
                                     b_ih1, b_hh1, b_fc, out, bar);
}
